// MoeLayer_32950989095494
// MI455X (gfx1250) — compile-verified, matrix operand order checked by reading, not run on hardware
//
#include <hip/hip_runtime.h>
#include <math.h>

#define D         1024
#define INTER     2816
#define NE        8
#define NTOK      8192        // b*l = 4*2048
#define ALPHA     0.01f
#define RES_ELEMS (NTOK * D)  // 8388608

// ---------------- CDNA5 WMMA types ----------------
typedef __attribute__((ext_vector_type(16))) __bf16 v16bf;
typedef __attribute__((ext_vector_type(8)))  float  v8f;

union Frag {
    v16bf    bf;
    uint4    q[2];
    unsigned u[8];
};

// f32 -> bf16 round-to-nearest-even (cold paths only)
__device__ __forceinline__ unsigned short f2bf(float f) {
    unsigned x = __float_as_uint(f);
    x += 0x7FFFu + ((x >> 16) & 1u);
    return (unsigned short)(x >> 16);
}
__device__ __forceinline__ unsigned pack2(float lo, float hi) {
    return (unsigned)f2bf(lo) | ((unsigned)f2bf(hi) << 16);
}
// single v_perm_b32 pack (truncate): {hi[31:16], lo[31:16]}
__device__ __forceinline__ unsigned packhi(float lo, float hi) {
    return __builtin_amdgcn_perm(__float_as_uint(hi), __float_as_uint(lo), 0x07060302u);
}

// K-offset of VGPR v within a 16x32 bf16 A/B fragment (ISA 7.12.2)
__device__ __forceinline__ int kp_of(int v, int half) {
    return ((v & 4) << 2) + (half << 3) + ((v & 3) << 1);
}

// A-fragment from LDS row (bf16 pairs). Pair dwords for v=0..3 are contiguous
// at (kk/2 + half*4), and for v=4..7 at (+8): two ds_load_b128.
__device__ __forceinline__ void load_afrag(const unsigned* row, int kk, int half, Frag& f) {
    const uint4* rp = (const uint4*)row;
    const int b = (kk >> 3) + half;
    f.q[0] = rp[b];
    f.q[1] = rp[b + 2];
}
// B-fragment from pre-tiled bf16 weights: lane's 8 dwords contiguous -> 2x b128
__device__ __forceinline__ void load_bfrag(const uint4* base, int tile, int lane, Frag& f) {
    const uint4* p = base + (size_t)tile * 64 + lane * 2;
    f.q[0] = p[0];
    f.q[1] = p[1];
}

__device__ __forceinline__ v8f wmma_bf16(const Frag& a, const Frag& b, v8f c) {
    return __builtin_amdgcn_wmma_f32_16x16x32_bf16(false, a.bf, false, b.bf,
                                                   (short)0, c, false, false);
}

// ---------------- kernel 0: zero results + counters ----------------
__global__ void moe_zero(float* __restrict__ results, int* __restrict__ cnt) {
    size_t i = (size_t)blockIdx.x * blockDim.x + threadIdx.x;
#pragma unroll
    for (int j = 0; j < 4; ++j) {
        size_t idx = i * 4 + j;
        if (idx < (size_t)RES_ELEMS) results[idx] = 0.0f;
    }
    if (blockIdx.x == 0 && threadIdx.x < NE) cnt[threadIdx.x] = 0;
}

// ------- kernel 0b: pre-convert f32 (K x N per expert) -> tiled bf16 -------
// dst layout: per expert, tiles (kblk = K/32) x (nblk = N/16), 256 dwords/tile,
// dword (tile, lane, v) = pack(W[kblk*32 + kp_of(v,lane>>4)][nblk*16+(lane&15)], W[+1][...])
__global__ __launch_bounds__(256)
void moe_cvt(const float* __restrict__ src, unsigned* __restrict__ dst, int K, int N) {
    const int e    = blockIdx.y;
    const int kblk = blockIdx.x;
    const int tid  = threadIdx.x;
    const int nblkTot = N >> 4;
    __shared__ float ldsT[32][129];

    const float* sp = src + (size_t)e * K * N + (size_t)kblk * 32 * N;
    unsigned*    dp = dst + (size_t)e * (K >> 5) * nblkTot * 256
                          + (size_t)kblk * nblkTot * 256;

    for (int n0 = 0; n0 < N; n0 += 128) {
        __syncthreads();
        for (int i = tid; i < 32 * 128; i += 256) {
            int r = i >> 7, c = i & 127;
            ldsT[r][c] = sp[(size_t)r * N + n0 + c];      // coalesced rows
        }
        __syncthreads();
        for (int w = tid; w < 2048; w += 256) {           // 8 n-tiles x 256 dwords
            int nb   = w >> 8;
            int lane = (w >> 3) & 31;
            int v    = w & 7;
            int kp   = kp_of(v, lane >> 4);
            int c    = nb * 16 + (lane & 15);
            unsigned pkd = pack2(ldsT[kp][c], ldsT[kp + 1][c]);
            int tile = (n0 >> 4) + nb;
            dp[(size_t)tile * 256 + lane * 8 + v] = pkd;  // coalesced writes
        }
    }
}

// ---------------- kernel 1: gating (one wave32 per token) ----------------
__global__ void moe_gate(const float* __restrict__ h,
                         const float* __restrict__ gate_w,
                         float* __restrict__ route,
                         int*   __restrict__ cnt,
                         int*   __restrict__ tokList,
                         float* __restrict__ tokW) {
    const int wave = threadIdx.x >> 5;
    const int lane = threadIdx.x & 31;
    const int t    = blockIdx.x * 8 + wave;
    if (t >= NTOK) return;

    float acc[NE];
#pragma unroll
    for (int e = 0; e < NE; ++e) acc[e] = 0.0f;

    const float* hp = h + (size_t)t * D;
    for (int k = lane; k < D; k += 32) {
        float hv = hp[k];
        const float* gw = gate_w + (size_t)k * NE;
#pragma unroll
        for (int e = 0; e < NE; ++e) acc[e] += hv * gw[e];
    }
#pragma unroll
    for (int e = 0; e < NE; ++e) {
#pragma unroll
        for (int off = 16; off > 0; off >>= 1)
            acc[e] += __shfl_xor(acc[e], off, 32);
    }

    if (lane == 0) {
        int e0 = 0; float l0 = acc[0];
        for (int e = 1; e < NE; ++e) if (acc[e] > l0) { l0 = acc[e]; e0 = e; }
        int e1 = -1; float l1 = -3.4e38f;
        for (int e = 0; e < NE; ++e) if (e != e0 && acc[e] > l1) { l1 = acc[e]; e1 = e; }

        float w0 = 1.0f / (1.0f + __expf(l1 - l0));
        float w1 = 1.0f - w0;

        float* rp = route + (size_t)t * NE;
#pragma unroll
        for (int e = 0; e < NE; ++e)
            rp[e] = (e == e0) ? w0 : ((e == e1) ? w1 : 0.0f);

        int p0 = atomicAdd(&cnt[e0], 1);
        tokList[e0 * NTOK + p0] = t;  tokW[e0 * NTOK + p0] = w0;
        int p1 = atomicAdd(&cnt[e1], 1);
        tokList[e1 * NTOK + p1] = t;  tokW[e1 * NTOK + p1] = w1;
    }
}

// ---------------- kernel 2: load-balancing loss (deterministic) ----------
__global__ void moe_loss(const float* __restrict__ route, float* __restrict__ loss) {
    __shared__ float part[256];
    const int tid = threadIdx.x;
    const int e   = tid & 7;
    float s = 0.0f;
    for (int t = tid >> 3; t < NTOK; t += 32)
        s += route[t * NE + e];
    part[tid] = s;
    __syncthreads();
    if (tid < 8) {
        float p = 0.0f;
        for (int j = 0; j < 32; ++j) p += part[e + 8 * j];
        p *= (1.0f / (float)NTOK);
        part[tid] = p * p;
    }
    __syncthreads();
    if (tid == 0) {
        float l = 0.0f;
        for (int e2 = 0; e2 < NE; ++e2) l += part[e2];
        *loss = ALPHA * l;
    }
}

// ---------------- kernel 3: fused expert FFN, 32-token tiles -------------
// block = 256 threads (8 waves); one (expert, 32-token tile) per block.
template <bool PRE>
__global__ __launch_bounds__(256)
void moe_ffn(const float* __restrict__ h,
             const float* __restrict__ wg,
             const float* __restrict__ wu,
             const float* __restrict__ wd,
             const unsigned* __restrict__ bwg,
             const unsigned* __restrict__ bwu,
             const unsigned* __restrict__ bwd,
             const int*   __restrict__ cnt,
             const int*   __restrict__ tokList,
             const float* __restrict__ tokW,
             float* __restrict__ results) {
    const int e     = blockIdx.y;
    const int tile  = blockIdx.x;
    const int count = cnt[e];
    if (tile * 32 >= count) return;

    __shared__ __attribute__((aligned(16))) unsigned hpk[32][516]; // bf16 pairs, 16B-aligned rows
    __shared__ __attribute__((aligned(16))) unsigned apk[32][68];  // act chunk 32x128 bf16
    __shared__ int   s_tok[32];
    __shared__ float s_wt[32];

    const int tid  = threadIdx.x;
    const int wave = tid >> 5;
    const int lane = tid & 31;
    const int half = lane >> 4;
    const int l15  = lane & 15;

    if (tid < 32) {
        int idx = tile * 32 + tid;
        if (idx < count) { s_tok[tid] = tokList[e * NTOK + idx]; s_wt[tid] = tokW[e * NTOK + idx]; }
        else             { s_tok[tid] = 0;                       s_wt[tid] = 0.0f; }
    }
    __syncthreads();

    for (int i = tid; i < 32 * (D / 2); i += 256) {
        int m = i >> 9, pj = i & 511;
        float2 hv = ((const float2*)(h + (size_t)s_tok[m] * D))[pj];
        hpk[m][pj] = pack2(hv.x, hv.y);
    }
    __syncthreads();

    const v8f vzero = {0.f, 0.f, 0.f, 0.f, 0.f, 0.f, 0.f, 0.f};
    v8f co[2][8];
#pragma unroll
    for (int mt = 0; mt < 2; ++mt)
#pragma unroll
        for (int t = 0; t < 8; ++t) co[mt][t] = vzero;

    const size_t estrideT = (size_t)D * INTER / 2;         // dwords per expert (tiled)
    const uint4* bwg_e = (const uint4*)(bwg + (size_t)e * estrideT);
    const uint4* bwu_e = (const uint4*)(bwu + (size_t)e * estrideT);
    const uint4* bwd_e = (const uint4*)(bwd + (size_t)e * estrideT);
    const size_t wbase = (size_t)e * D * INTER;            // f32 fallback bases
    const size_t dbase = (size_t)e * INTER * D;

    for (int ic = 0; ic < INTER; ic += 128) {              // 22 chunks
        // ---- phase 1: g,u for this wave's 16 inter-columns, 2 M-tiles ----
        v8f cg[2] = {vzero, vzero}, cu[2] = {vzero, vzero};
        for (int kk = 0; kk < D; kk += 32) {
            Frag a0, a1, bg, bu;
            load_afrag(&hpk[l15][0],      kk, half, a0);
            load_afrag(&hpk[16 + l15][0], kk, half, a1);
            if constexpr (PRE) {
                const int t1 = (kk >> 5) * (INTER >> 4) + (ic >> 4) + wave;
                load_bfrag(bwg_e, t1, lane, bg);
                load_bfrag(bwu_e, t1, lane, bu);
            } else {
                const int ncol = ic + wave * 16 + l15;
                const float* wgp = wg + wbase + ncol;
                const float* wup = wu + wbase + ncol;
#pragma unroll
                for (int v = 0; v < 8; ++v) {
                    size_t ro = (size_t)(kk + kp_of(v, half)) * INTER;
                    bg.u[v] = packhi(wgp[ro], wgp[ro + INTER]);
                    bu.u[v] = packhi(wup[ro], wup[ro + INTER]);
                }
            }
            cg[0] = wmma_bf16(a0, bg, cg[0]);
            cu[0] = wmma_bf16(a0, bu, cu[0]);
            cg[1] = wmma_bf16(a1, bg, cg[1]);
            cu[1] = wmma_bf16(a1, bu, cu[1]);
        }

        __syncthreads();                                   // prior phase-2 readers done
        {
            unsigned short* ap0 = (unsigned short*)&apk[0][0];
            const int rs = 2 * 68;                          // ushorts per padded row
#pragma unroll
            for (int mt = 0; mt < 2; ++mt)
#pragma unroll
                for (int v = 0; v < 8; ++v) {
                    int m = mt * 16 + v + 8 * half;
                    float g = cg[mt][v];
                    float s = g * __builtin_amdgcn_rcpf(1.0f + __expf(-g)) * cu[mt][v];
                    ap0[m * rs + wave * 16 + l15] = f2bf(s);
                }
        }
        __syncthreads();

        // ---- phase 2: out += act(32x128) @ wd_chunk(128x1024), 128 cols/wave ----
        const int nb = wave * 128;
        for (int kk = 0; kk < 128; kk += 32) {
            Frag a0, a1;
            load_afrag(&apk[l15][0],      kk, half, a0);
            load_afrag(&apk[16 + l15][0], kk, half, a1);
#pragma unroll
            for (int t = 0; t < 8; ++t) {
                Frag bw;
                if constexpr (PRE) {
                    const int t2 = ((ic + kk) >> 5) * (D >> 4) + wave * 8 + t;
                    load_bfrag(bwd_e, t2, lane, bw);
                } else {
                    const float* wdp = wd + dbase + (size_t)(ic + kk) * D + nb + t * 16 + l15;
#pragma unroll
                    for (int v = 0; v < 8; ++v) {
                        int kp = kp_of(v, half);
                        bw.u[v] = packhi(wdp[(size_t)kp * D], wdp[(size_t)(kp + 1) * D]);
                    }
                }
                co[0][t] = wmma_bf16(a0, bw, co[0][t]);
                co[1][t] = wmma_bf16(a1, bw, co[1][t]);
            }
        }
    }

    // ---- combine: scale by routing weight; 2 commutative f32 atomics/token ----
#pragma unroll
    for (int mt = 0; mt < 2; ++mt)
#pragma unroll
        for (int t = 0; t < 8; ++t) {
            const int n = wave * 128 + t * 16 + l15;
#pragma unroll
            for (int v = 0; v < 8; ++v) {
                const int m = mt * 16 + v + 8 * half;
                atomicAdd(&results[(size_t)s_tok[m] * D + n], co[mt][t][v] * s_wt[m]);
            }
        }
}

// ---------------- launcher ----------------
extern "C" void kernel_launch(void* const* d_in, const int* in_sizes, int n_in,
                              void* d_out, int out_size, void* d_ws, size_t ws_size,
                              hipStream_t stream) {
    const float* h      = (const float*)d_in[0];
    const float* gate_w = (const float*)d_in[1];
    const float* wg     = (const float*)d_in[2];
    const float* wu     = (const float*)d_in[3];
    const float* wd     = (const float*)d_in[4];

    float* results = (float*)d_out;
    float* loss    = results + RES_ELEMS;
    float* route   = loss + 1;

    int*   cnt     = (int*)d_ws;
    int*   tokList = cnt + NE;
    float* tokW    = (float*)(tokList + NE * NTOK);

    const size_t lists_bytes = (size_t)1 << 20;                 // 1 MB for lists
    const size_t mat_dwords  = (size_t)NE * D * INTER / 2;      // per weight matrix
    const size_t need        = lists_bytes + 3 * mat_dwords * 4;
    const bool   pre         = ws_size >= need;

    unsigned* bwg = (unsigned*)((char*)d_ws + lists_bytes);
    unsigned* bwu = bwg + mat_dwords;
    unsigned* bwd = bwu + mat_dwords;

    moe_zero<<<RES_ELEMS / 1024, 256, 0, stream>>>(results, cnt);
    if (pre) {
        moe_cvt<<<dim3(D / 32, NE),     256, 0, stream>>>(wg, bwg, D, INTER);
        moe_cvt<<<dim3(D / 32, NE),     256, 0, stream>>>(wu, bwu, D, INTER);
        moe_cvt<<<dim3(INTER / 32, NE), 256, 0, stream>>>(wd, bwd, INTER, D);
    }
    moe_gate<<<NTOK / 8, 256, 0, stream>>>(h, gate_w, route, cnt, tokList, tokW);
    moe_loss<<<1, 256, 0, stream>>>(route, loss);
    if (pre)
        moe_ffn<true><<<dim3(256, NE), 256, 0, stream>>>(h, wg, wu, wd, bwg, bwu, bwd,
                                                         cnt, tokList, tokW, results);
    else
        moe_ffn<false><<<dim3(256, NE), 256, 0, stream>>>(h, wg, wu, wd, bwg, bwu, bwd,
                                                          cnt, tokList, tokW, results);
}
